// Attn_20813411516524
// MI455X (gfx1250) — matrix operand order repaired and checked by reading, not compiled, not run on hardware
//
#include <hip/hip_runtime.h>
#include <hip/hip_bf16.h>

// ---- types for WMMA fragments (CDNA5 wave32) ----
typedef __attribute__((ext_vector_type(16))) __bf16 v16bf;
typedef __attribute__((ext_vector_type(8)))  __bf16 v8bf;
typedef __attribute__((ext_vector_type(8)))  float  v8f;

#define WMMA_BF16(a, b, c) \
  __builtin_amdgcn_wmma_f32_16x16x32_bf16(false, (a), false, (b), (short)0, (c), false, false)

__device__ __forceinline__ unsigned short f2bf(float f) {
  unsigned int u = __builtin_bit_cast(unsigned int, f);
  u += 0x7FFFu + ((u >> 16) & 1u);            // round-to-nearest-even
  return (unsigned short)(u >> 16);
}

// A-fragment (16-bit, 16x32): lane = row (lane&15); halves 0..15 = K kb..kb+15,
// kb = (lane>>4)*16. Caller points p at row*lda + kb.
__device__ __forceinline__ v16bf mk_afrag(const unsigned short* p) {
  v8bf lo = *(const v8bf*)(p);
  v8bf hi = *(const v8bf*)(p + 8);
  v16bf r;
#pragma unroll
  for (int i = 0; i < 8; ++i) { r[i] = lo[i]; r[8 + i] = hi[i]; }
  return r;
}

// B-fragment (16-bit, 32x16): lane = col (lane&15); halves = K k2..k2+15,
// k2 = (lane>>4)*16 -> 16 contiguous bf16 (32B aligned) in a K-major layout.
__device__ __forceinline__ v16bf ld_bfrag(const unsigned short* p) {
  return *(const v16bf*)(p);
}

// ---------------- conversion kernels ----------------
__global__ void cvt_x_kernel(const float* __restrict__ x, unsigned short* __restrict__ xb, int n) {
  int i = blockIdx.x * blockDim.x + threadIdx.x;
  if (i < n) xb[i] = f2bf(x[i]);
}

// Coalesced 32x32 LDS-tiled transpose: wT[f*K + k] = bf16(w[k*Nn + f])
__global__ __launch_bounds__(256) void cvt_wT_kernel(const float* __restrict__ w,
                                                     unsigned short* __restrict__ wT,
                                                     int K, int Nn) {
  __shared__ unsigned short tile[32][33];
  int kt = blockIdx.x << 5, ft = blockIdx.y << 5;
  int tx = threadIdx.x & 31, ty = threadIdx.x >> 5;   // ty = 0..7
#pragma unroll
  for (int i = 0; i < 32; i += 8)
    tile[ty + i][tx] = f2bf(w[(size_t)(kt + ty + i) * Nn + ft + tx]);   // coalesced over f
  __syncthreads();
#pragma unroll
  for (int i = 0; i < 32; i += 8)
    wT[(size_t)(ft + ty + i) * K + kt + tx] = tile[tx][ty + i];         // coalesced over k
}

// ---------------- QKV projection GEMM ----------------
// M=8192 (B*N), K=768, Nout=2304. 64x64 tile per wave: 4 A-frags x 4 B-frags = 16 WMMAs
// per k-step from 8 fragment loads (register-level reuse, ~1 b128 per WMMA).
// Q stored (b,h,n,d) pre-scaled by 1/sqrt(d); K stored (b,h,n,d); V stored transposed (b,h,d,n).
__global__ __launch_bounds__(256) void qkv_gemm_kernel(
    const unsigned short* __restrict__ xb, const unsigned short* __restrict__ wT,
    const float* __restrict__ bias, unsigned short* __restrict__ Qo,
    unsigned short* __restrict__ Ko, unsigned short* __restrict__ Vt) {
  const int K = 768;
  const int NTILE = 2304 / 64;  // 36
  int wave = blockIdx.x * 8 + (threadIdx.x >> 5);
  int lane = threadIdx.x & 31;
  int tm = wave / NTILE;
  int tn = wave - tm * NTILE;
  int m0 = tm << 6, n0 = tn << 6;

  const unsigned short* ab = xb + (size_t)(m0 + (lane & 15)) * K + ((lane >> 4) << 4);
  const unsigned short* bb = wT + (size_t)(n0 + (lane & 15)) * K + ((lane >> 4) << 4);

  v8f acc[4][4] = {};
  for (int k0 = 0; k0 < K; k0 += 32) {
    v16bf a0 = mk_afrag(ab + k0);
    v16bf a1 = mk_afrag(ab + (size_t)16 * K + k0);
    v16bf a2 = mk_afrag(ab + (size_t)32 * K + k0);
    v16bf a3 = mk_afrag(ab + (size_t)48 * K + k0);
#pragma unroll
    for (int ni = 0; ni < 4; ++ni) {
      v16bf b = ld_bfrag(bb + (size_t)(ni * 16) * K + k0);
      acc[0][ni] = WMMA_BF16(a0, b, acc[0][ni]);
      acc[1][ni] = WMMA_BF16(a1, b, acc[1][ni]);
      acc[2][ni] = WMMA_BF16(a2, b, acc[2][ni]);
      acc[3][ni] = WMMA_BF16(a3, b, acc[3][ni]);
    }
  }

  float bv[4];
#pragma unroll
  for (int ni = 0; ni < 4; ++ni) bv[ni] = bias[n0 + ni * 16 + (lane & 15)];

  // whole 64-wide tile lies in one of Q/K/V and one head (boundaries are multiples of 64)
  int rup = (lane >> 4) << 3;
  if (n0 < 768) {
    int hh = n0 >> 6;
#pragma unroll
    for (int mi = 0; mi < 4; ++mi)
#pragma unroll
      for (int r = 0; r < 8; ++r) {
        int m = m0 + mi * 16 + r + rup;
        int bbx = m >> 11, n = m & 2047;
        size_t rowo = (((size_t)(bbx * 12 + hh)) * 2048 + n) * 64 + (lane & 15);
#pragma unroll
        for (int ni = 0; ni < 4; ++ni)
          Qo[rowo + ni * 16] = f2bf((acc[mi][ni][r] + bv[ni]) * 0.125f);  // 1/sqrt(64)
      }
  } else if (n0 < 1536) {
    int hh = (n0 - 768) >> 6;
#pragma unroll
    for (int mi = 0; mi < 4; ++mi)
#pragma unroll
      for (int r = 0; r < 8; ++r) {
        int m = m0 + mi * 16 + r + rup;
        int bbx = m >> 11, n = m & 2047;
        size_t rowo = (((size_t)(bbx * 12 + hh)) * 2048 + n) * 64 + (lane & 15);
#pragma unroll
        for (int ni = 0; ni < 4; ++ni)
          Ko[rowo + ni * 16] = f2bf(acc[mi][ni][r] + bv[ni]);
      }
  } else {
    int hh = (n0 - 1536) >> 6;
#pragma unroll
    for (int mi = 0; mi < 4; ++mi)
#pragma unroll
      for (int r = 0; r < 8; ++r) {
        int m = m0 + mi * 16 + r + rup;
        int bbx = m >> 11, n = m & 2047;
        size_t base = ((size_t)(bbx * 12 + hh)) * 64 * 2048 + n;
#pragma unroll
        for (int ni = 0; ni < 4; ++ni)
          Vt[base + (size_t)(ni * 16 + (lane & 15)) * 2048] = f2bf(acc[mi][ni][r] + bv[ni]);
      }
  }
}

// ---------------- flash attention ----------------
// One wave: 16 query rows of one (b,h); 64-key blocks (16 WMMAs/iter); online softmax with
// one max-reduction per 64 keys and the row-sum reduction deferred to the epilogue
// (corr is uniform across the 16-lane row group, so per-lane partial l is exact).
__global__ __launch_bounds__(256) void attn_fwd_kernel(
    const unsigned short* __restrict__ Q, const unsigned short* __restrict__ Kd,
    const unsigned short* __restrict__ Vt, unsigned short* __restrict__ AO) {
  __shared__ __align__(32) unsigned short ldsP[8][16 * 64];  // per-wave 16x64 bf16 P tile
  int wid  = threadIdx.x >> 5;
  int lane = threadIdx.x & 31;
  int wave = blockIdx.x * 8 + wid;
  int qblk = wave & 127;           // N/16 = 128
  int bh   = wave >> 7;            // 0..47
  const unsigned short* Qp = Q  + (size_t)bh * 2048 * 64;
  const unsigned short* Kp = Kd + (size_t)bh * 2048 * 64;
  const unsigned short* Vp = Vt + (size_t)bh * 64 * 2048;
  int q0 = qblk << 4;
  int rA = lane & 15;
  int kb = (lane >> 4) << 4;       // 0 / 16
  int k2 = (lane >> 4) << 4;       // 0 / 16

  v16bf qa0 = mk_afrag(Qp + (size_t)(q0 + rA) * 64 + kb);
  v16bf qa1 = mk_afrag(Qp + (size_t)(q0 + rA) * 64 + kb + 32);

  float mrow[8], lpart[8];
  v8f o0 = {}, o1 = {}, o2 = {}, o3 = {};
#pragma unroll
  for (int r = 0; r < 8; ++r) { mrow[r] = -3.0e38f; lpart[r] = 0.0f; }
  unsigned short* Pw = &ldsP[wid][0];

  for (int j0 = 0; j0 < 2048; j0 += 64) {
    // S tile: 16 queries x 64 keys = 4 col-tiles x 2 k-chunks = 8 WMMAs
    v8f s[4];
#pragma unroll
    for (int t = 0; t < 4; ++t) {
      const unsigned short* kc = Kp + (size_t)(j0 + t * 16 + rA) * 64 + k2;
      v8f z = {};
      s[t] = WMMA_BF16(qa0, ld_bfrag(kc), z);
      s[t] = WMMA_BF16(qa1, ld_bfrag(kc + 32), s[t]);
    }
    __builtin_prefetch(Kp + (size_t)(j0 + 64 + rA) * 64 + k2, 0, 3);
    __builtin_prefetch(Vp + (size_t)(lane & 15) * 2048 + j0 + 64 + k2, 0, 3);

    // online softmax (one reduction per 64 keys)
#pragma unroll
    for (int r = 0; r < 8; ++r) {
      float vmax = fmaxf(fmaxf(s[0][r], s[1][r]), fmaxf(s[2][r], s[3][r]));
      vmax = fmaxf(vmax, __shfl_xor(vmax, 8, 16));
      vmax = fmaxf(vmax, __shfl_xor(vmax, 4, 16));
      vmax = fmaxf(vmax, __shfl_xor(vmax, 2, 16));
      vmax = fmaxf(vmax, __shfl_xor(vmax, 1, 16));
      float mn = fmaxf(mrow[r], vmax);
      float cr = __expf(mrow[r] - mn);
      float p0 = __expf(s[0][r] - mn);
      float p1 = __expf(s[1][r] - mn);
      float p2 = __expf(s[2][r] - mn);
      float p3 = __expf(s[3][r] - mn);
      lpart[r] = lpart[r] * cr + (p0 + p1) + (p2 + p3);  // per-lane partial sum
      mrow[r] = mn;
      int prow = r + ((lane >> 4) << 3);
      Pw[prow * 64 +  0 + (lane & 15)] = f2bf(p0);
      Pw[prow * 64 + 16 + (lane & 15)] = f2bf(p1);
      Pw[prow * 64 + 32 + (lane & 15)] = f2bf(p2);
      Pw[prow * 64 + 48 + (lane & 15)] = f2bf(p3);
      o0[r] *= cr; o1[r] *= cr; o2[r] *= cr; o3[r] *= cr;
    }
    asm volatile("s_wait_dscnt 0x0" ::: "memory");     // P tile visible for cross-lane reload

    // O += P(16x64) * V(64x64): 2 P A-frags x 4 d-chunks = 8 WMMAs
    v16bf pa0 = mk_afrag(Pw + rA * 64 + kb);
    v16bf pa1 = mk_afrag(Pw + rA * 64 + kb + 32);
#pragma unroll
    for (int c = 0; c < 4; ++c) {
      const unsigned short* vb = Vp + (size_t)(c * 16 + (lane & 15)) * 2048 + j0 + k2;
      v8f* oc = (c == 0) ? &o0 : (c == 1) ? &o1 : (c == 2) ? &o2 : &o3;
      *oc = WMMA_BF16(pa0, ld_bfrag(vb), *oc);
      *oc = WMMA_BF16(pa1, ld_bfrag(vb + 32), *oc);
    }
  }

  // deferred row-sum reduction, normalize, store (b, n, h*64+dcol) bf16
  int bb = bh / 12, hh = bh % 12;
#pragma unroll
  for (int r = 0; r < 8; ++r) {
    float l = lpart[r];
    l += __shfl_xor(l, 8, 16);
    l += __shfl_xor(l, 4, 16);
    l += __shfl_xor(l, 2, 16);
    l += __shfl_xor(l, 1, 16);
    float inv = 1.0f / l;
    int n = q0 + r + ((lane >> 4) << 3);
    size_t off = ((size_t)(bb * 2048 + n)) * 768 + hh * 64 + (lane & 15);
    AO[off +  0] = f2bf(o0[r] * inv);
    AO[off + 16] = f2bf(o1[r] * inv);
    AO[off + 32] = f2bf(o2[r] * inv);
    AO[off + 48] = f2bf(o3[r] * inv);
  }
}

// ---------------- output projection GEMM (fp32 out + bias) ----------------
// 64x32 tile per wave: 4 A-frags x 2 B-frags = 8 WMMAs per k-step.
__global__ __launch_bounds__(256) void proj_gemm_kernel(
    const unsigned short* __restrict__ A, const unsigned short* __restrict__ wT,
    const float* __restrict__ bias, float* __restrict__ out) {
  const int K = 768;
  const int NTILE = 768 / 32;  // 24
  int wave = blockIdx.x * 8 + (threadIdx.x >> 5);
  int lane = threadIdx.x & 31;
  int tm = wave / NTILE;
  int tn = wave - tm * NTILE;
  int m0 = tm << 6, n0 = tn << 5;

  const unsigned short* ab = A  + (size_t)(m0 + (lane & 15)) * K + ((lane >> 4) << 4);
  const unsigned short* bb = wT + (size_t)(n0 + (lane & 15)) * K + ((lane >> 4) << 4);

  v8f acc[4][2] = {};
  for (int k0 = 0; k0 < K; k0 += 32) {
    v16bf a0 = mk_afrag(ab + k0);
    v16bf a1 = mk_afrag(ab + (size_t)16 * K + k0);
    v16bf a2 = mk_afrag(ab + (size_t)32 * K + k0);
    v16bf a3 = mk_afrag(ab + (size_t)48 * K + k0);
#pragma unroll
    for (int ni = 0; ni < 2; ++ni) {
      v16bf b = ld_bfrag(bb + (size_t)(ni * 16) * K + k0);
      acc[0][ni] = WMMA_BF16(a0, b, acc[0][ni]);
      acc[1][ni] = WMMA_BF16(a1, b, acc[1][ni]);
      acc[2][ni] = WMMA_BF16(a2, b, acc[2][ni]);
      acc[3][ni] = WMMA_BF16(a3, b, acc[3][ni]);
    }
  }

  float bv0 = bias[n0 + (lane & 15)];
  float bv1 = bias[n0 + 16 + (lane & 15)];
#pragma unroll
  for (int mi = 0; mi < 4; ++mi)
#pragma unroll
    for (int r = 0; r < 8; ++r) {
      int m = m0 + mi * 16 + r + ((lane >> 4) << 3);
      size_t rowo = (size_t)m * 768 + n0 + (lane & 15);
      out[rowo]      = acc[mi][0][r] + bv0;
      out[rowo + 16] = acc[mi][1][r] + bv1;
    }
}

// ---------------- host-side launch ----------------
extern "C" void kernel_launch(void* const* d_in, const int* in_sizes, int n_in,
                              void* d_out, int out_size, void* d_ws, size_t ws_size,
                              hipStream_t stream) {
  (void)in_sizes; (void)n_in; (void)out_size; (void)ws_size;
  const float* x     = (const float*)d_in[0];
  const float* Wqkv  = (const float*)d_in[1];
  const float* bqkv  = (const float*)d_in[2];
  const float* Wproj = (const float*)d_in[3];
  const float* bproj = (const float*)d_in[4];
  float* out = (float*)d_out;

  // workspace partition (bf16 stored as ushort)
  unsigned short* ws = (unsigned short*)d_ws;
  const size_t SZ = (size_t)8192 * 768;            // 6,291,456 elements
  unsigned short* xb     = ws;
  unsigned short* wqkvT  = xb + SZ;
  unsigned short* wprojT = wqkvT + (size_t)2304 * 768;
  unsigned short* Qd     = wprojT + (size_t)768 * 768;
  unsigned short* Kd     = Qd + SZ;
  unsigned short* Vtd    = Kd + SZ;
  unsigned short* AO     = Vtd + SZ;               // total ~67.6 MB

  int nX = 8192 * 768;
  cvt_x_kernel<<<(nX + 255) / 256, 256, 0, stream>>>(x, xb, nX);
  cvt_wT_kernel<<<dim3(768 / 32, 2304 / 32), 256, 0, stream>>>(Wqkv, wqkvT, 768, 2304);
  cvt_wT_kernel<<<dim3(768 / 32, 768 / 32), 256, 0, stream>>>(Wproj, wprojT, 768, 768);

  qkv_gemm_kernel<<<(128 * 36) / 8, 256, 0, stream>>>(xb, wqkvT, bqkv, Qd, Kd, Vtd);
  attn_fwd_kernel<<<6144 / 8, 256, 0, stream>>>(Qd, Kd, Vtd, AO);
  proj_gemm_kernel<<<(128 * 24) / 8, 256, 0, stream>>>(AO, wprojT, bproj, out);
}
